// STConvBlock_13262859010734
// MI455X (gfx1250) — compile-verified, matrix operand order checked by reading, not run on hardware
//
#include <hip/hip_runtime.h>
#include <math.h>

// Problem constants (match reference)
#define BB 4
#define NN 4096
#define TT 16
#define CC 64
#define EE 32768
#define TWOC 128
#define TILE (TT*CC)          // 1024 floats per (b,n) node tile
#define ROW  (BB*TT*CC)       // 4096 floats per graph-node row (node-major)
#define NNODES (BB*NN)        // 16384
#define NELEM (BB*NN*TT*CC)   // 16,777,216
#define EPSV 1e-5f

typedef __attribute__((ext_vector_type(16))) __bf16 bf16x16;
typedef __attribute__((ext_vector_type(8)))  float  f32x8;

typedef union {
    bf16x16 v;
    unsigned u[8];
    uint4 q[2];
} fragU;

// float -> bf16 bits, round-to-nearest-even
__device__ __forceinline__ unsigned bf16bits(float f) {
    union { float f; unsigned u; } v; v.f = f;
    return (v.u + 0x7FFFu + ((v.u >> 16) & 1u)) >> 16;
}
// pack two floats into one dword of two bf16 (lo = a, hi = b)
__device__ __forceinline__ unsigned pack2bf(float a, float b) {
    return (bf16bits(a) & 0xFFFFu) | (bf16bits(b) << 16);
}

__device__ __forceinline__ f32x8 wmma_bf16(bf16x16 a, bf16x16 b, f32x8 c) {
    // D = A(16x32 bf16) * B(32x16 bf16) + C(16x16 f32)
    return __builtin_amdgcn_wmma_f32_16x16x32_bf16(false, a, false, b, (short)0, c,
                                                   false, false);
}

// K-index of the first element in VGPR j for this lane (CDNA5 16-bit A/B layout):
// VGPR0..3 : lanes0-15 K=2j..2j+1, lanes16-31 K=2j+8..  ; VGPR4..7 : +16
__device__ __forceinline__ int frag_kk(int j, int h) {
    return (j < 4) ? (2*j + 8*h) : (16 + 2*(j-4) + 8*h);
}

// A fragment, folded-conv GEMM. xs = 18x64 bf16 tile (row0 = t=-1 zero pad).
// K-block [32s,32s+32) has constant tap = s>>1, so per-lane data is two
// 16B-aligned contiguous chunks -> 2x ds_load_b128, already in WMMA pair layout.
__device__ __forceinline__ bf16x16 afrag_conv(const __bf16* xs, int lane, int s) {
    fragU f;
    int m = lane & 15, h = lane >> 4;
    const unsigned* row = (const unsigned*)(xs + (m + (s >> 1))*64 + (s & 1)*32);
    f.q[0] = *(const uint4*)(row + 4*h);        // elements 8h .. 8h+7
    f.q[1] = *(const uint4*)(row + 8 + 4*h);    // elements 16+8h .. 16+8h+7
    return f.v;
}

// A fragment, plain 16x64 tile (K=64, s in 0..1)
__device__ __forceinline__ bf16x16 afrag_flat(const __bf16* xs, int lane, int s) {
    fragU f;
    int m = lane & 15, h = lane >> 4;
    const unsigned* row = (const unsigned*)(xs + m*64 + 32*s);
    f.q[0] = *(const uint4*)(row + 4*h);
    f.q[1] = *(const uint4*)(row + 8 + 4*h);
    return f.v;
}

// B fragment from pre-packed LDS: Wpk[((s*NT+nt)*32+lane)*8 .. +7] -> 2x ds_load_b128
__device__ __forceinline__ bf16x16 bfrag_pk(const unsigned* Wpk, int lane, int s,
                                            int nt, int NT) {
    fragU f;
    const unsigned* p = Wpk + (size_t)(((s*NT) + nt)*32 + lane)*8;
    f.q[0] = *(const uint4*)(p);
    f.q[1] = *(const uint4*)(p + 4);
    return f.v;
}

// Stage conv weight W[o][c][k] (o<128, c<64, k<3) into per-lane fragment order.
// kc = k*64+c; fragment dword j of lane ln, tile nt, step s covers kc=kk,kk+1.
__device__ __forceinline__ void stage_wconv(unsigned* Wpk, const float* __restrict__ W,
                                            int tid) {
    for (int i = tid; i < 6*8*32*8; i += 256) {
        int j  = i & 7;
        int ln = (i >> 3) & 31;
        int nt = (i >> 8) & 7;
        int s  = i >> 11;
        int h  = ln >> 4;
        int kk = frag_kk(j, h) + 32*s;          // even; kk,kk+1 share tap
        int n  = (ln & 15) + 16*nt;             // output channel
        int k0 = kk >> 6, c0 = kk & 63;
        float w0 = W[n*192 + c0*3 + k0];
        float w1 = W[n*192 + (c0 + 1)*3 + k0];
        Wpk[i] = pack2bf(w0, w1);
    }
}

// ---------------- Stage 1: temporal GLU #1 (+ReLU) ----------------
__global__ __launch_bounds__(256) void k_glu1(const float* __restrict__ x,
                                              const float* __restrict__ W1,
                                              const float* __restrict__ b1,
                                              float* __restrict__ h1) {
    __shared__ __align__(16) unsigned Wpk[6*8*32*8];      // 48 KB
    __shared__ __align__(16) __bf16 Xsh[8][18*64];        // 18 KB
    int tid = threadIdx.x, lane = tid & 31, w = tid >> 5;

    stage_wconv(Wpk, W1, tid);

    int node = blockIdx.x * 8 + w;
    const float4* xp4 = (const float4*)(x + (size_t)node * TILE);
    __bf16* xs = Xsh[w];
    {
        uint2* dst = (uint2*)(xs + 64);                   // rows t=0..15
        for (int i = lane; i < 256; i += 32) {
            float4 v = xp4[i];
            uint2 d; d.x = pack2bf(v.x, v.y); d.y = pack2bf(v.z, v.w);
            dst[i] = d;
        }
        ((unsigned*)xs)[lane] = 0u;                       // pad row t=-1
        ((unsigned*)(xs + 17*64))[lane] = 0u;             // pad row t=16
    }
    __syncthreads();

    f32x8 acc[8] = {};
#pragma unroll
    for (int s = 0; s < 6; ++s) {
        bf16x16 a = afrag_conv(xs, lane, s);
#pragma unroll
        for (int nt = 0; nt < 8; ++nt)
            acc[nt] = wmma_bf16(a, bfrag_pk(Wpk, lane, s, nt, 8), acc[nt]);
    }

    int m0 = (lane >> 4) * 8, c0 = lane & 15;
    float* hp = h1 + (size_t)node * TILE;
#pragma unroll
    for (int nt = 0; nt < 4; ++nt) {
        int c = c0 + 16*nt;
        float bv = b1[c], bgt = b1[64 + c];
#pragma unroll
        for (int r = 0; r < 8; ++r) {
            float v = acc[nt][r] + bv;
            float g = acc[nt + 4][r] + bgt;
            float o = v * (1.f / (1.f + __expf(-g)));
            hp[(r + m0)*64 + c] = fmaxf(o, 0.f);
        }
    }
}

// ---------------- Stage 2: GCN pointwise matmul ----------------
// y[...,d] = sum_c h1[...,c]*Wg[d][c] + bg[d]; written NODE-MAJOR: y[(n*B+b)*TILE]
__global__ __launch_bounds__(256) void k_gcn(const float* __restrict__ h1,
                                             const float* __restrict__ Wg,
                                             const float* __restrict__ bg,
                                             float* __restrict__ y) {
    __shared__ __align__(16) unsigned Wpk[2*4*32*8];      // 8 KB
    __shared__ __align__(16) __bf16 Xsh[8][16*64];        // 16 KB
    int tid = threadIdx.x, lane = tid & 31, w = tid >> 5;

    for (int i = tid; i < 2*4*32*8; i += 256) {
        int j  = i & 7;
        int ln = (i >> 3) & 31;
        int nt = (i >> 8) & 3;
        int s  = i >> 10;
        int h  = ln >> 4;
        int kk = frag_kk(j, h) + 32*s;                    // c index
        int d  = (ln & 15) + 16*nt;
        Wpk[i] = pack2bf(Wg[d*64 + kk], Wg[d*64 + kk + 1]);
    }
    int node = blockIdx.x * 8 + w;
    const float4* hp4 = (const float4*)(h1 + (size_t)node * TILE);
    __bf16* xs = Xsh[w];
    {
        uint2* dst = (uint2*)xs;
        for (int i = lane; i < 256; i += 32) {
            float4 v = hp4[i];
            uint2 d; d.x = pack2bf(v.x, v.y); d.y = pack2bf(v.z, v.w);
            dst[i] = d;
        }
    }
    __syncthreads();

    f32x8 acc[4] = {};
#pragma unroll
    for (int s = 0; s < 2; ++s) {
        bf16x16 a = afrag_flat(xs, lane, s);
#pragma unroll
        for (int nt = 0; nt < 4; ++nt)
            acc[nt] = wmma_bf16(a, bfrag_pk(Wpk, lane, s, nt, 4), acc[nt]);
    }

    int bidx = node >> 12, nidx = node & (NN - 1);        // node = b*N + n
    float* yp = y + ((size_t)nidx * BB + bidx) * TILE;
    int m0 = (lane >> 4) * 8, c0 = lane & 15;
#pragma unroll
    for (int nt = 0; nt < 4; ++nt) {
        int d = c0 + 16*nt;
        float bd = bg[d];
#pragma unroll
        for (int r = 0; r < 8; ++r)
            yp[(r + m0)*64 + d] = acc[nt][r] + bd;
    }
}

// ---------------- Stage 3: edge-weighted segment sum ----------------
__global__ __launch_bounds__(256) void k_fill0(float4* __restrict__ p, int n4) {
    int i = blockIdx.x * 256 + threadIdx.x;
    if (i < n4) p[i] = make_float4(0.f, 0.f, 0.f, 0.f);
}

// agg[src] += w_e * y[dst]  (rows of ROW=4096 floats; y is 67MB -> L2 resident)
__global__ __launch_bounds__(256) void k_edges(const int* __restrict__ esrc,
                                               const int* __restrict__ edst,
                                               const float* __restrict__ ew,
                                               const float* __restrict__ y,
                                               float* __restrict__ agg) {
    int e = blockIdx.x;
    int s = esrc[e], d = edst[e];
    float w = ew[e];
    const float4* yp4 = (const float4*)(y + (size_t)d * ROW);
    float* ap = agg + (size_t)s * ROW;
    for (int i = threadIdx.x; i < ROW/4; i += 256) {
        float4 v = yp4[i];
        atomicAdd(ap + 4*i + 0, w * v.x);
        atomicAdd(ap + 4*i + 1, w * v.y);
        atomicAdd(ap + 4*i + 2, w * v.z);
        atomicAdd(ap + 4*i + 3, w * v.w);
    }
}

// ---------------- Stage 4: temporal GLU #2 + residual + channel stats ----------------
// input agg is node-major (flat f = n*B + b); applies relu(agg) on load.
__global__ __launch_bounds__(256) void k_glu2(const float* __restrict__ agg,
                                              const float* __restrict__ W2,
                                              const float* __restrict__ b2,
                                              const float* __restrict__ x,
                                              float* __restrict__ out,
                                              float* __restrict__ stats) {
    __shared__ __align__(16) unsigned Wpk[6*8*32*8];      // 48 KB
    __shared__ __align__(16) __bf16 Xsh[8][18*64];        // 18 KB
    __shared__ float sstat[128];
    int tid = threadIdx.x, lane = tid & 31, w = tid >> 5;

    if (tid < 128) sstat[tid] = 0.f;
    stage_wconv(Wpk, W2, tid);

    int f = blockIdx.x * 8 + w;                           // f = n*B + b (node-major)
    const float4* ap4 = (const float4*)(agg + (size_t)f * TILE);
    __bf16* xs = Xsh[w];
    {
        uint2* dst = (uint2*)(xs + 64);
        for (int i = lane; i < 256; i += 32) {
            float4 v = ap4[i];
            uint2 d;
            d.x = pack2bf(fmaxf(v.x, 0.f), fmaxf(v.y, 0.f));   // relu on load
            d.y = pack2bf(fmaxf(v.z, 0.f), fmaxf(v.w, 0.f));
            dst[i] = d;
        }
        ((unsigned*)xs)[lane] = 0u;
        ((unsigned*)(xs + 17*64))[lane] = 0u;
    }
    __syncthreads();

    f32x8 acc[8] = {};
#pragma unroll
    for (int s = 0; s < 6; ++s) {
        bf16x16 a = afrag_conv(xs, lane, s);
#pragma unroll
        for (int nt = 0; nt < 8; ++nt)
            acc[nt] = wmma_bf16(a, bfrag_pk(Wpk, lane, s, nt, 8), acc[nt]);
    }

    int nidx = f >> 2, bidx = f & 3;                      // B = 4
    size_t base = ((size_t)bidx * NN + nidx) * TILE;
    const float* rp = x + base;
    float* op = out + base;
    int m0 = (lane >> 4) * 8, c0 = lane & 15;
#pragma unroll
    for (int nt = 0; nt < 4; ++nt) {
        int c = c0 + 16*nt;
        float bv = b2[c], bgt = b2[64 + c];
        float psum = 0.f, psq = 0.f;
#pragma unroll
        for (int r = 0; r < 8; ++r) {
            float v = acc[nt][r] + bv;
            float g = acc[nt + 4][r] + bgt;
            float o = v * (1.f / (1.f + __expf(-g)));
            o += rp[(r + m0)*64 + c];                     // residual
            op[(r + m0)*64 + c] = o;
            psum += o; psq += o * o;
        }
        atomicAdd(&sstat[c], psum);                       // ds_add_f32
        atomicAdd(&sstat[64 + c], psq);
    }
    __syncthreads();
    if (tid < 64) {
        atomicAdd(&stats[tid], sstat[tid]);
        atomicAdd(&stats[64 + tid], sstat[64 + tid]);
    }
}

// ---------------- Stage 5: finalize per-channel mean / rsqrt(var+eps) ----------------
__global__ void k_stats(float* __restrict__ stats) {
    int c = threadIdx.x;
    if (c < 64) {
        const float M = (float)((size_t)BB * NN * TT);
        float mean = stats[c] / M;
        float var  = stats[64 + c] / M - mean * mean;
        stats[128 + c] = mean;
        stats[192 + c] = rsqrtf(var + EPSV);
    }
}

// ---------------- Stage 6: normalize + affine + relu (in place, float4) ----------------
__global__ __launch_bounds__(256) void k_norm(float* __restrict__ out,
                                              const float* __restrict__ stats,
                                              const float* __restrict__ gamma,
                                              const float* __restrict__ beta) {
    int i4 = blockIdx.x * 256 + threadIdx.x;              // one float4
    float4* o4 = (float4*)out;
    float4 v = o4[i4];
    int c = (i4 << 2) & 63;                               // 4 consecutive channels
    float m0 = stats[128 + c],     i0 = stats[192 + c];
    float m1 = stats[128 + c + 1], i1 = stats[192 + c + 1];
    float m2 = stats[128 + c + 2], i2 = stats[192 + c + 2];
    float m3 = stats[128 + c + 3], i3 = stats[192 + c + 3];
    v.x = fmaxf((v.x - m0) * i0 * gamma[c]     + beta[c],     0.f);
    v.y = fmaxf((v.y - m1) * i1 * gamma[c + 1] + beta[c + 1], 0.f);
    v.z = fmaxf((v.z - m2) * i2 * gamma[c + 2] + beta[c + 2], 0.f);
    v.w = fmaxf((v.w - m3) * i3 * gamma[c + 3] + beta[c + 3], 0.f);
    o4[i4] = v;
}

extern "C" void kernel_launch(void* const* d_in, const int* in_sizes, int n_in,
                              void* d_out, int out_size, void* d_ws, size_t ws_size,
                              hipStream_t stream) {
    (void)in_sizes; (void)n_in; (void)out_size; (void)ws_size;
    const float* x    = (const float*)d_in[0];
    const int*   esrc = (const int*)d_in[1];
    const int*   edst = (const int*)d_in[2];
    const float* ew   = (const float*)d_in[3];
    const float* W1   = (const float*)d_in[4];
    const float* b1   = (const float*)d_in[5];
    const float* Wg   = (const float*)d_in[6];
    const float* bg   = (const float*)d_in[7];
    const float* W2   = (const float*)d_in[8];
    const float* b2   = (const float*)d_in[9];
    const float* gamma = (const float*)d_in[10];
    const float* beta  = (const float*)d_in[11];
    float* out = (float*)d_out;

    // Workspace: buf0 (h1, later reused as agg) | buf1 (y, node-major) | stats (256 f)
    float* buf0  = (float*)d_ws;
    float* buf1  = buf0 + (size_t)NELEM;
    float* stats = buf1 + (size_t)NELEM;

    k_glu1 <<<NNODES / 8, 256, 0, stream>>>(x, W1, b1, buf0);
    k_gcn  <<<NNODES / 8, 256, 0, stream>>>(buf0, Wg, bg, buf1);
    k_fill0<<<NELEM / 1024, 256, 0, stream>>>((float4*)buf0, NELEM / 4);  // zero agg
    k_fill0<<<1, 256, 0, stream>>>((float4*)stats, 64);                    // zero stats
    k_edges<<<EE, 256, 0, stream>>>(esrc, edst, ew, buf1, buf0);
    k_glu2 <<<NNODES / 8, 256, 0, stream>>>(buf0, W2, b2, x, out, stats);
    k_stats<<<1, 64, 0, stream>>>(stats);
    k_norm <<<NELEM / 1024, 256, 0, stream>>>(out, stats, gamma, beta);
}
